// SMILESDecoder_76330158784601
// MI455X (gfx1250) — matrix operand order repaired and checked by reading, not compiled, not run on hardware
//
#include <hip/hip_runtime.h>
#include <hip/hip_bf16.h>

typedef __attribute__((ext_vector_type(16))) __bf16 v16bf;
typedef __attribute__((ext_vector_type(8)))  float  v8f;

#define B_      1024
#define LATENT_ 2048
#define H_      512
#define V_      128
#define L_      256
#define G4_     2048   // 4*H

// workspace layout (bytes)
#define WS_W1B    0u        // bf16 [H][LATENT]      2 MB
#define WS_WIHB   2097152u  // bf16 [4H][H]          2 MB
#define WS_WSUMB  4194304u  // bf16 [4H][H]          2 MB
#define WS_W2B    6291456u  // bf16 [V][H]           128 KB
#define WS_BSUM   6422528u  // f32  [4H]             8 KB

static __device__ __forceinline__ float sigm_(float x) {
  return 1.0f / (1.0f + __expf(-x));
}

static __device__ __forceinline__ unsigned pk2(float a, float b) {
  unsigned short ua = __builtin_bit_cast(unsigned short, (__bf16)a);
  unsigned short ub = __builtin_bit_cast(unsigned short, (__bf16)b);
  return (unsigned)ua | ((unsigned)ub << 16);
}

// Store 32 consecutive-K f32 values (row M, K-chunk ch) into LDS in the
// contiguous WMMA 16-bit A-fragment layout (matches B operand loads):
//   lane M   : K 0..15  (bytes 0..31)
//   lane M+16: K 16..31 (bytes 0..31)
static __device__ __forceinline__ void storeAfrag(unsigned* abuf_u32, int M, int ch,
                                                  const float* v) {
  unsigned* lo = abuf_u32 + ch * 256 + M * 8;          // 1 KB/chunk, 32 B/lane
  unsigned* hi = abuf_u32 + ch * 256 + (M + 16) * 8;
  lo[0] = pk2(v[0], v[1]);   lo[1] = pk2(v[2], v[3]);
  lo[2] = pk2(v[4], v[5]);   lo[3] = pk2(v[6], v[7]);
  lo[4] = pk2(v[8], v[9]);   lo[5] = pk2(v[10], v[11]);
  lo[6] = pk2(v[12], v[13]); lo[7] = pk2(v[14], v[15]);
  hi[0] = pk2(v[16], v[17]); hi[1] = pk2(v[18], v[19]);
  hi[2] = pk2(v[20], v[21]); hi[3] = pk2(v[22], v[23]);
  hi[4] = pk2(v[24], v[25]); hi[5] = pk2(v[26], v[27]);
  hi[6] = pk2(v[28], v[29]); hi[7] = pk2(v[30], v[31]);
}

// fc2 projection for one step using register-resident A fragments (h state).
// out[bTile*16 + M, step, col] = h @ W2^T + b2, one 16-col N-tile per wave.
static __device__ __forceinline__ void fc2_project(const v16bf* areg,
                                                   const __bf16* __restrict__ W2b,
                                                   const float* __restrict__ b2,
                                                   float* __restrict__ out,
                                                   int bTile, int step,
                                                   int wave, int laneN, int laneH) {
  int col = wave * 16 + laneN;
  v8f acc = {};
  const __bf16* bp = W2b + (size_t)col * H_ + laneH * 16;
  #pragma unroll
  for (int k = 0; k < 16; ++k) {
    v16bf b = *(const v16bf*)(bp + k * 32);
    acc = __builtin_amdgcn_wmma_f32_16x16x32_bf16(false, areg[k], false, b,
                                                  (short)0, acc, false, false);
  }
  float bias = b2[col];
  int mo = laneH * 8;
  #pragma unroll
  for (int g = 0; g < 8; ++g)
    out[((size_t)(bTile * 16 + mo + g)) * L_ * V_ + (size_t)step * V_ + col] =
        acc[g] + bias;
}

// One-time weight conversion: W1/W_ih -> bf16, Wsum = W_ih + W_hh (bf16),
// W2 -> bf16, bsum = b_ih + b_hh (f32).
__global__ void __launch_bounds__(256)
prep_kernel(const float* __restrict__ W1, const float* __restrict__ Wih,
            const float* __restrict__ Whh, const float* __restrict__ W2,
            const float* __restrict__ bih, const float* __restrict__ bhh,
            __bf16* __restrict__ W1b, __bf16* __restrict__ Wihb,
            __bf16* __restrict__ Wsumb, __bf16* __restrict__ W2b,
            float* __restrict__ bsum) {
  int i = blockIdx.x * 256 + threadIdx.x;
  if (i < H_ * LATENT_) {           // == 4H * H == 1048576 for both shapes
    W1b[i]   = (__bf16)W1[i];
    Wihb[i]  = (__bf16)Wih[i];
    Wsumb[i] = (__bf16)(Wih[i] + Whh[i]);
  }
  if (i < V_ * H_)  W2b[i]  = (__bf16)W2[i];
  if (i < G4_)      bsum[i] = bih[i] + bhh[i];
}

// Fused fc1 -> 256-step LSTM -> fc2. One workgroup (256 thr, 8 waves) per
// 16-row batch tile. LDS: gates f32 [16][2048] (128 KB) + A-frag buffer
// (64 chunks * 1 KB, loop uses first 16 chunks). A operand is hoisted into
// 128 VGPRs per lane each step; c state lives in registers.
__global__ void __launch_bounds__(256)
lstm_fused_kernel(const float* __restrict__ fp,     // [B, LATENT]
                  const float* __restrict__ b1,     // [H]
                  const float* __restrict__ b2,     // [V]
                  const __bf16* __restrict__ W1b,   // [H, LATENT]
                  const __bf16* __restrict__ Wihb,  // [4H, H]
                  const __bf16* __restrict__ Wsumb, // [4H, H]
                  const __bf16* __restrict__ W2b,   // [V, H]
                  const float* __restrict__ bsum,   // [4H]
                  float* __restrict__ out)          // [B, L, V]
{
  extern __shared__ char smem[];
  float*   gates   = (float*)smem;                     // [16][2048]
  __bf16*  Abuf    = (__bf16*)(smem + 16 * G4_ * 4);   // A-fragment buffer
  unsigned* Abuf32 = (unsigned*)Abuf;

  const int tid   = threadIdx.x;
  const int wave  = tid >> 5;
  const int lane  = tid & 31;
  const int laneN = lane & 15;      // N / column within tile
  const int laneH = lane >> 4;      // half-wave selector
  const int bTile = blockIdx.x;     // 64 tiles of 16 rows
  const int rowM  = tid >> 4;       // 0..15  (elementwise row ownership)
  const int chOwn = tid & 15;       // owned K-chunk for h/x0 writes

  // ---- Load fingerprints tile into A-fragment layout (64 chunks) ----
  {
    const float* src = fp + (size_t)(bTile * 16 + rowM) * LATENT_;
    for (int j = 0; j < 4; ++j) {
      int ch = chOwn + 16 * j;
      float v[32];
      const float4* s4 = (const float4*)(src + ch * 32);
      #pragma unroll
      for (int q = 0; q < 8; ++q) {
        float4 t = s4[q];
        v[4*q] = t.x; v[4*q+1] = t.y; v[4*q+2] = t.z; v[4*q+3] = t.w;
      }
      storeAfrag(Abuf32, rowM, ch, v);
    }
  }
  __syncthreads();

  // ---- fc1: x0[16,512] = fpTile @ W1^T  (N: 32 tiles -> 4/wave, K: 64 chunks) ----
  for (int j = 0; j < 4; ++j) {
    int col = (wave * 4 + j) * 16 + laneN;
    v8f acc = {};
    const __bf16* bp = W1b + (size_t)col * LATENT_ + laneH * 16;
    #pragma unroll 4
    for (int k = 0; k < 64; ++k) {
      v16bf a = *(const v16bf*)(Abuf + k * 512 + lane * 16);
      v16bf b = *(const v16bf*)(bp + k * 32);
      acc = __builtin_amdgcn_wmma_f32_16x16x32_bf16(false, a, false, b,
                                                    (short)0, acc, false, false);
    }
    int mo = laneH * 8;
    #pragma unroll
    for (int g = 0; g < 8; ++g) gates[(mo + g) * 512 + col] = acc[g];  // x0 scratch
  }
  __syncthreads();

  // ---- x0 + b1 -> A-fragment chunks 0..15 ----
  {
    float v[32];
    #pragma unroll
    for (int q = 0; q < 32; ++q)
      v[q] = gates[rowM * 512 + chOwn * 32 + q] + b1[chOwn * 32 + q];
    storeAfrag(Abuf32, rowM, chOwn, v);
  }

  float creg[32];
  #pragma unroll
  for (int q = 0; q < 32; ++q) creg[q] = 0.0f;

  v16bf areg[16];   // whole A operand (16x512 bf16 tile striped over the wave)

  for (int s = 0; s < L_; ++s) {
    __syncthreads();  // A-buffer (x_s) ready; gates buffer free

    // Hoist A fragments into registers: 32 ds_load_b128 per wave per step.
    #pragma unroll
    for (int k = 0; k < 16; ++k)
      areg[k] = *(const v16bf*)(Abuf + k * 512 + lane * 16);

    // fc2 for the previous step reuses the just-loaded A (= h_{s-1}).
    if (s > 0)
      fc2_project(areg, W2b, b2, out, bTile, s - 1, wave, laneN, laneH);

    // gates[16,2048] = A @ Wp^T + bsum   (N: 128 tiles -> 16/wave, K: 16 chunks)
    // s==0: A=x0, Wp=W_ih (h0==0).  s>=1: A=h, Wp=W_ih+W_hh (x==h).
    const __bf16* Wp = (s == 0) ? Wihb : Wsumb;
    for (int j = 0; j < 16; ++j) {
      int col = (wave * 16 + j) * 16 + laneN;
      v8f acc = {};
      const __bf16* bp = Wp + (size_t)col * H_ + laneH * 16;
      #pragma unroll
      for (int k = 0; k < 16; ++k) {
        v16bf b = *(const v16bf*)(bp + k * 32);
        acc = __builtin_amdgcn_wmma_f32_16x16x32_bf16(false, areg[k], false, b,
                                                      (short)0, acc, false, false);
      }
      float bias = bsum[col];
      int mo = laneH * 8;
      #pragma unroll
      for (int g = 0; g < 8; ++g) gates[(mo + g) * G4_ + col] = acc[g] + bias;
    }
    __syncthreads();  // gates visible; all A-register ds reads fenced

    // elementwise LSTM cell (PyTorch gate order i,f,g,o); h_s -> A-frag buffer
    {
      float hv[32];
      const float* gr = gates + rowM * G4_ + chOwn * 32;
      #pragma unroll
      for (int q = 0; q < 32; ++q) {
        float gi = gr[q];
        float gf = gr[512 + q];
        float gg = gr[1024 + q];
        float go = gr[1536 + q];
        float cn = sigm_(gf) * creg[q] + sigm_(gi) * tanhf(gg);
        creg[q] = cn;
        hv[q]   = sigm_(go) * tanhf(cn);
      }
      storeAfrag(Abuf32, rowM, chOwn, hv);
    }
  }

  // epilogue: fc2 for the final step (h_255)
  __syncthreads();
  #pragma unroll
  for (int k = 0; k < 16; ++k)
    areg[k] = *(const v16bf*)(Abuf + k * 512 + lane * 16);
  fc2_project(areg, W2b, b2, out, bTile, L_ - 1, wave, laneN, laneH);
}

extern "C" void kernel_launch(void* const* d_in, const int* in_sizes, int n_in,
                              void* d_out, int out_size, void* d_ws, size_t ws_size,
                              hipStream_t stream) {
  const float* fp  = (const float*)d_in[0];
  const float* W1  = (const float*)d_in[1];
  const float* b1  = (const float*)d_in[2];
  const float* Wih = (const float*)d_in[3];
  const float* Whh = (const float*)d_in[4];
  const float* bih = (const float*)d_in[5];
  const float* bhh = (const float*)d_in[6];
  const float* W2  = (const float*)d_in[7];
  const float* b2  = (const float*)d_in[8];
  float* out = (float*)d_out;

  char* ws = (char*)d_ws;
  __bf16* W1b   = (__bf16*)(ws + WS_W1B);
  __bf16* Wihb  = (__bf16*)(ws + WS_WIHB);
  __bf16* Wsumb = (__bf16*)(ws + WS_WSUMB);
  __bf16* W2b   = (__bf16*)(ws + WS_W2B);
  float*  bsum  = (float*)(ws + WS_BSUM);

  prep_kernel<<<(H_ * LATENT_ + 255) / 256, 256, 0, stream>>>(
      W1, Wih, Whh, W2, bih, bhh, W1b, Wihb, Wsumb, W2b, bsum);

  const size_t ldsBytes = (size_t)16 * G4_ * 4 + (size_t)64 * 512 * 2;  // 192 KB
  lstm_fused_kernel<<<B_ / 16, 256, ldsBytes, stream>>>(
      fp, b1, b2, W1b, Wihb, Wsumb, W2b, bsum, out);
}
